// AttentionLayer_52201032515789
// MI455X (gfx1250) — compile-verified, matrix operand order checked by reading, not run on hardware
//
#include <hip/hip_runtime.h>
#include <hip/hip_bf16.h>

// ---------------------------------------------------------------------------
// AttentionLayer on MI455X (gfx1250): bf16 WMMA pipeline
//   x(B,S,DM) @ w_qkv(DM,H,F) -> RoPE -> flash attention (ALiBi+causal)
//   -> attn @ w_out(H,DH,DM) -> out(B,S,DM) f32
// GEMMs: 64x256 block tiles, async global->LDS staging (ASYNCcnt),
//        fragments from LDS (ds_load_b128), v_wmma_f32_16x16x32_bf16.
// Attention: 64-row Q blocks, 4 waves share async-staged K/V chunks in LDS
//        (4x less K/V traffic vs per-wave streaming).
// ---------------------------------------------------------------------------

typedef __bf16 bf16_t;
typedef bf16_t v16bf __attribute__((ext_vector_type(16)));
typedef bf16_t v8bf  __attribute__((ext_vector_type(8)));
typedef float  v8f   __attribute__((ext_vector_type(8)));

#define WMMA_BF16(a, b, c) \
  __builtin_amdgcn_wmma_f32_16x16x32_bf16(false, (a), false, (b), (short)0, (c), false, false)

#define ASYNC_LDS_B128(lds_addr, gaddr) \
  asm volatile("global_load_async_to_lds_b128 %0, %1, off" \
               :: "v"(lds_addr), "v"(gaddr) : "memory")

#define WAIT_ASYNC0() asm volatile("s_wait_asynccnt 0x0" ::: "memory")

// Problem constants
constexpr int B_  = 2;
constexpr int S_  = 2048;
constexpr int DM_ = 1024;
constexpr int H_  = 16;
constexpr int F_  = 192;
constexpr int DH_ = 64;
constexpr int M_  = B_ * S_;      // 4096 rows
constexpr int NQKV_ = H_ * F_;    // 3072
constexpr int NAV_  = H_ * DH_;   // 1024

// Workspace layout (bytes). Total 72 MiB.
constexpr size_t OFF_XB  = 0;                         // x bf16          8 MiB
constexpr size_t OFF_WQT = 8ull  << 20;               // w_qkv^T bf16    6 MiB
constexpr size_t OFF_WOT = 14ull << 20;               // w_out^T bf16    2 MiB
constexpr size_t OFF_QKV = 16ull << 20;               // qkv bf16       24 MiB
constexpr size_t OFF_Q   = 40ull << 20;               // q (B,H,S,DH)    8 MiB
constexpr size_t OFF_K   = 48ull << 20;               // k (B,H,S,DH)    8 MiB
constexpr size_t OFF_VT  = 56ull << 20;               // v^T (B,H,DH,S)  8 MiB
constexpr size_t OFF_AT  = 64ull << 20;               // attn bf16       8 MiB

// ---------------------------------------------------------------------------
// Fragment helpers. CDNA5 ISA §7.12.2, 16-bit operands, wave32:
//   A (16x32): lane L holds row M=L&15; half=L>>4 selects K ranges
//              elems 0..7  -> K = half*8 + {0..7}
//              elems 8..15 -> K = 16 + half*8 + {0..7}
//   B (32x16): same pattern with N=L&15 when loaded from an N-major (B^T) buffer.
//   C/D (16x16 f32): elem r at lane L -> (M = r + 8*(L>>4), N = L&15)
// Both 8-element chunks are contiguous 16-byte (b128) loads.
// ---------------------------------------------------------------------------
__device__ __forceinline__ v16bf load_frag16(const bf16_t* p) {
  union { v16bf v; v8bf h[2]; } u;
  u.h[0] = *(const v8bf*)(p);
  u.h[1] = *(const v8bf*)(p + 16);
  return u.v;
}

// ---------------------------------------------------------------------------
// Pack kernels
// ---------------------------------------------------------------------------
__global__ void cvt_f32_to_bf16(const float* __restrict__ src, bf16_t* __restrict__ dst, int n) {
  int i = blockIdx.x * blockDim.x + threadIdx.x;
  if (i < n) dst[i] = (bf16_t)src[i];
}

// src: row-major K x N f32 ; dst: row-major N x K bf16 (i.e. B^T for frag loads)
__global__ void transpose_cvt_bf16(const float* __restrict__ src, bf16_t* __restrict__ dst,
                                   int K, int N) {
  int i = blockIdx.x * blockDim.x + threadIdx.x;
  if (i < K * N) {
    int kk = i / N;
    int n  = i - kk * N;
    dst[(size_t)n * K + kk] = (bf16_t)src[i];
  }
}

// ---------------------------------------------------------------------------
// GEMM: C(MxN) = A(MxK) * B(KxN) + bias[n]; A row-major bf16, Bt = B^T (NxK).
// Block: 512 threads = 16 waves, 4(M) x 4(N). Block tile 64 x 256, K-slab 64
// staged in LDS per step via global_load_async_to_lds_b128. All 10 fragments
// of a slab are loaded before the 8 WMMAs so DS latency can be overlapped
// with partial s_wait_dscnt instead of wait-0 serialization.
// ---------------------------------------------------------------------------
constexpr int BM = 64, BN = 256, BK = 64;

__global__ __launch_bounds__(512)
void gemm_bf16_wmma_lds(const bf16_t* __restrict__ A, const bf16_t* __restrict__ Bt,
                        void* __restrict__ Cout, const float* __restrict__ bias,
                        int M, int N, int K, int out_bf16) {
  __shared__ __align__(16) bf16_t sA[BM * BK];   // row m: 64 k contiguous (128B/row)
  __shared__ __align__(16) bf16_t sB[BN * BK];   // row n: 64 k contiguous

  const int tid  = threadIdx.x;
  const int lane = tid & 31;
  const int wave = tid >> 5;
  const int wm = wave & 3;           // M sub-tile 0..3
  const int wn = wave >> 2;          // N sub-tile 0..3
  const int m0 = blockIdx.y * BM;
  const int n0 = blockIdx.x * BN;
  const int nl = lane & 15, half = lane >> 4;

  // LDS byte addresses: generic shared pointer low 32 bits == LDS offset.
  const unsigned ldsA = (unsigned)(uintptr_t)(void*)sA;
  const unsigned ldsB = (unsigned)(uintptr_t)(void*)sB;

  v8f acc0 = {}, acc1 = {}, acc2 = {}, acc3 = {};

  for (int kb = 0; kb < K; kb += BK) {
    __syncthreads();   // prior LDS consumption complete before overwrite

    // ---- stage A slab: 8 KiB = 512 x 16B chunks, one per thread
    {
      const int o = tid << 4;
      const int mrow = o >> 7, koff = (o & 127) >> 1;
      ASYNC_LDS_B128(ldsA + (unsigned)o,
          (unsigned long long)(uintptr_t)(A + (size_t)(m0 + mrow) * K + kb + koff));
    }
    // ---- stage B slab: 32 KiB = 2048 chunks, four per thread
    #pragma unroll
    for (int j = 0; j < 4; ++j) {
      const int o = (tid + (j << 9)) << 4;
      const int nrow = o >> 7, koff = (o & 127) >> 1;
      ASYNC_LDS_B128(ldsB + (unsigned)o,
          (unsigned long long)(uintptr_t)(Bt + (size_t)(n0 + nrow) * K + kb + koff));
    }

    // ---- overlap: prefetch next K-slab into caches while DMA completes
    if (kb + BK < K) {
      __builtin_prefetch(Bt + (size_t)(n0 + (tid >> 1)) * K + kb + BK + ((tid & 1) << 5), 0, 1);
      if (tid < 64)
        __builtin_prefetch(A + (size_t)(m0 + tid) * K + kb + BK, 0, 1);
    }

    WAIT_ASYNC0();
    __syncthreads();   // staged slabs visible to all waves

    // ---- load ALL fragments for the 64-K slab, then run 8 WMMAs
    const bf16_t* ap = sA + (size_t)(wm * 16 + nl) * BK + half * 8;
    const bf16_t* bp = sB + (size_t)(wn * 64 + nl) * BK + half * 8;
    v16bf a0  = load_frag16(ap);            // K chunk 0..31
    v16bf a1  = load_frag16(ap + 32);       // K chunk 32..63
    v16bf b00 = load_frag16(bp);
    v16bf b10 = load_frag16(bp + 16 * BK);
    v16bf b20 = load_frag16(bp + 32 * BK);
    v16bf b30 = load_frag16(bp + 48 * BK);
    v16bf b01 = load_frag16(bp + 32);
    v16bf b11 = load_frag16(bp + 16 * BK + 32);
    v16bf b21 = load_frag16(bp + 32 * BK + 32);
    v16bf b31 = load_frag16(bp + 48 * BK + 32);
    acc0 = WMMA_BF16(a0, b00, acc0);
    acc1 = WMMA_BF16(a0, b10, acc1);
    acc2 = WMMA_BF16(a0, b20, acc2);
    acc3 = WMMA_BF16(a0, b30, acc3);
    acc0 = WMMA_BF16(a1, b01, acc0);
    acc1 = WMMA_BF16(a1, b11, acc1);
    acc2 = WMMA_BF16(a1, b21, acc2);
    acc3 = WMMA_BF16(a1, b31, acc3);
  }

  // ---- epilogue: bias + store (f32 or bf16)
  const int mw = m0 + wm * 16;
  const int nwb = n0 + wn * 64;
  const int rbase = half << 3;
  #pragma unroll
  for (int t = 0; t < 4; ++t) {
    v8f acc = (t == 0) ? acc0 : (t == 1) ? acc1 : (t == 2) ? acc2 : acc3;
    const int n = nwb + (t << 4) + nl;
    const float bv = bias[n];
    #pragma unroll
    for (int r = 0; r < 8; ++r) {
      const size_t idx = (size_t)(mw + rbase + r) * N + n;
      if (out_bf16) ((bf16_t*)Cout)[idx] = (bf16_t)(acc[r] + bv);
      else          ((float*)Cout)[idx]  = acc[r] + bv;
    }
  }
}

// ---------------------------------------------------------------------------
// RoPE + layout scatter.
// qkv bf16 (B,S,H,F) -> q,k bf16 (B,H,S,DH) (rotated), v^T bf16 (B,H,DH,S)
// ---------------------------------------------------------------------------
__global__ void rope_scatter(const bf16_t* __restrict__ qkv, bf16_t* __restrict__ q,
                             bf16_t* __restrict__ k, bf16_t* __restrict__ vt) {
  const int idx = blockIdx.x * blockDim.x + threadIdx.x;
  if (idx >= B_ * S_ * H_ * DH_) return;
  const int dh = idx & (DH_ - 1);
  const int h  = (idx >> 6) & (H_ - 1);
  const int s  = (idx >> 10) & (S_ - 1);
  const int b  = idx >> 21;

  const bf16_t* base = qkv + (((size_t)(b * S_ + s)) * H_ + h) * F_;
  const int j = dh & 31;                               // freq index (sin/cos concat)
  const float inv_freq = __powf(10000.0f, -(float)(2 * j) * (1.0f / (float)DH_));
  float sn, cs;
  __sincosf((float)s * inv_freq, &sn, &cs);

  const float qv = (float)base[dh];
  const float kv = (float)base[DH_ + dh];
  const float vv = (float)base[2 * DH_ + dh];
  const float qp = (dh < 32) ? -(float)base[dh + 32] : (float)base[dh - 32];
  const float kp = (dh < 32) ? -(float)base[DH_ + dh + 32] : (float)base[DH_ + dh - 32];

  const size_t o = (((size_t)(b * H_ + h)) * S_ + s) * DH_ + dh;
  q[o] = (bf16_t)(qv * cs + qp * sn);
  k[o] = (bf16_t)(kv * cs + kp * sn);
  vt[(((size_t)(b * H_ + h)) * DH_ + dh) * S_ + s] = (bf16_t)vv;
}

// ---------------------------------------------------------------------------
// Flash attention, causal + ALiBi.
// Block = 128 threads (4 waves), 64 Q rows per block (16 per wave).
// Per 32-KV chunk: K (32x64, 4 KiB) and V^T (64x32, 4 KiB) are async-DMA'd
// into LDS once and consumed by all 4 waves -> 4x less K/V global traffic.
// Online softmax in f32 with per-wave LDS scratch. Fully-masked chunks for
// upper waves are mathematically inert (exp(-1e9-m)=0, corr=1).
// ---------------------------------------------------------------------------
__global__ __launch_bounds__(128)
void flash_attn_wmma(const bf16_t* __restrict__ Qm, const bf16_t* __restrict__ Km,
                     const bf16_t* __restrict__ Vt, bf16_t* __restrict__ attnb) {
  __shared__ __align__(16) bf16_t sK[32 * 64];   // kv-row major, 64 dh (128B/row)
  __shared__ __align__(16) bf16_t sV[64 * 32];   // dh-row major, 32 kv (64B/row)
  __shared__ float sP[4][16][32];                // per-wave P tiles
  __shared__ float s_max[4][16], s_sum[4][16], s_scale[4][16];

  const int tid  = threadIdx.x;
  const int lane = tid & 31;
  const int wave = tid >> 5;                     // q sub-tile 0..3
  const int h = blockIdx.y, b = blockIdx.z;
  const int m0 = blockIdx.x * 64;
  const int mw = m0 + wave * 16;                 // this wave's q rows
  const int nl = lane & 15, half = lane >> 4, rbase = half << 3;

  const bf16_t* qbase = Qm + (size_t)(b * H_ + h) * S_ * DH_;
  const bf16_t* kbase = Km + (size_t)(b * H_ + h) * S_ * DH_;
  const bf16_t* vbase = Vt + (size_t)(b * H_ + h) * DH_ * S_;

  const unsigned ldsK = (unsigned)(uintptr_t)(void*)sK;
  const unsigned ldsV = (unsigned)(uintptr_t)(void*)sV;

  // Q tile 16x64 -> two A-frags (K = 0..31, 32..63)
  const bf16_t* qrow = qbase + (size_t)(mw + nl) * DH_ + half * 8;
  const v16bf qa0 = load_frag16(qrow);
  const v16bf qa1 = load_frag16(qrow + 32);

  v8f o0 = {}, o1 = {}, o2 = {}, o3 = {};
  if (lane < 16) { s_max[wave][lane] = -3.0e38f; s_sum[wave][lane] = 0.0f; }

  const float scale  = 0.07216878364870323f;                 // 1/sqrt(F=192)
  const float slope8 = exp2f(-0.5f * (float)(h + 1)) * 8.0f; // ALiBi slope * MAX_BIAS

  for (int kv = 0; kv < m0 + 64; kv += 32) {
    __syncthreads();   // prior consumption of sK/sV/sP complete

    // ---- async-stage K chunk: 256 x 16B chunks (2 per thread)
    #pragma unroll
    for (int j = 0; j < 2; ++j) {
      const int o = (tid + (j << 7)) << 4;
      const int krow = o >> 7, koff = (o & 127) >> 1;
      ASYNC_LDS_B128(ldsK + (unsigned)o,
          (unsigned long long)(uintptr_t)(kbase + (size_t)(kv + krow) * DH_ + koff));
    }
    // ---- async-stage V chunk: 256 x 16B chunks (2 per thread)
    #pragma unroll
    for (int j = 0; j < 2; ++j) {
      const int o = (tid + (j << 7)) << 4;
      const int vrow = o >> 6, voff = (o & 63) >> 1;
      ASYNC_LDS_B128(ldsV + (unsigned)o,
          (unsigned long long)(uintptr_t)(vbase + (size_t)vrow * S_ + kv + voff));
    }
    WAIT_ASYNC0();
    __syncthreads();

    // ---- scores: 16 x 32 chunk, K=DH=64 -> 4 WMMAs (K frags from LDS)
    const bf16_t* kp = sK + (size_t)nl * 64 + half * 8;
    v16bf kb00 = load_frag16(kp);                // cols kv..kv+15,  dh 0..31
    v16bf kb01 = load_frag16(kp + 32);           // cols kv..kv+15,  dh 32..63
    v16bf kb10 = load_frag16(kp + 16 * 64);      // cols kv+16..+31, dh 0..31
    v16bf kb11 = load_frag16(kp + 16 * 64 + 32); // cols kv+16..+31, dh 32..63
    v8f sc0 = {}, sc1 = {};
    sc0 = WMMA_BF16(qa0, kb00, sc0);
    sc0 = WMMA_BF16(qa1, kb01, sc0);
    sc1 = WMMA_BF16(qa0, kb10, sc1);
    sc1 = WMMA_BF16(qa1, kb11, sc1);

    // ---- scale + ALiBi + causal mask, stash to per-wave LDS tile
    #pragma unroll
    for (int r = 0; r < 8; ++r) {
      const int m = mw + rbase + r;
      const int na = kv + nl, nb = na + 16;
      float v0 = sc0[r] * scale + slope8 * (float)(na - m);
      float v1 = sc1[r] * scale + slope8 * (float)(nb - m);
      if (na > m) v0 = -1.0e9f;
      if (nb > m) v1 = -1.0e9f;
      sP[wave][rbase + r][nl] = v0;
      sP[wave][rbase + r][16 + nl] = v1;
    }
    __syncthreads();

    // ---- online softmax: lanes 0..15 each own one row of their wave's tile
    if (lane < 16) {
      float mx = s_max[wave][lane], rowmax = mx;
      #pragma unroll
      for (int j2 = 0; j2 < 32; ++j2) rowmax = fmaxf(rowmax, sP[wave][lane][j2]);
      const float corr = __expf(mx - rowmax);
      float sum = s_sum[wave][lane] * corr;
      #pragma unroll
      for (int j2 = 0; j2 < 32; ++j2) {
        const float p = __expf(sP[wave][lane][j2] - rowmax);
        sP[wave][lane][j2] = p;
        sum += p;
      }
      s_max[wave][lane] = rowmax; s_sum[wave][lane] = sum; s_scale[wave][lane] = corr;
    }
    __syncthreads();

    // ---- rescale running O accumulators by per-row correction
    #pragma unroll
    for (int r = 0; r < 8; ++r) {
      const float c = s_scale[wave][rbase + r];
      o0[r] *= c; o1[r] *= c; o2[r] *= c; o3[r] *= c;
    }

    // ---- P as bf16 A-frag (16x32) straight from LDS
    v16bf pa;
    #pragma unroll
    for (int i2 = 0; i2 < 8; ++i2) {
      pa[i2]     = (bf16_t)sP[wave][nl][(half << 3) + i2];
      pa[8 + i2] = (bf16_t)sP[wave][nl][16 + (half << 3) + i2];
    }

    // ---- O += P * V : K = 32 kv positions, N = 64 dh (V frags from LDS)
    const bf16_t* vp = sV + (size_t)nl * 32 + half * 8;
    v16bf vb0 = load_frag16(vp);
    v16bf vb1 = load_frag16(vp + 16 * 32);
    v16bf vb2 = load_frag16(vp + 32 * 32);
    v16bf vb3 = load_frag16(vp + 48 * 32);
    o0 = WMMA_BF16(pa, vb0, o0);
    o1 = WMMA_BF16(pa, vb1, o1);
    o2 = WMMA_BF16(pa, vb2, o2);
    o3 = WMMA_BF16(pa, vb3, o3);
  }

  // ---- normalize and store attn (B,S,H,DH) bf16
  #pragma unroll
  for (int r = 0; r < 8; ++r) {
    const int m = mw + rbase + r;
    const float inv = 1.0f / s_sum[wave][rbase + r];
    bf16_t* op = attnb + (((size_t)(b * S_ + m)) * H_ + h) * DH_ + nl;
    op[0]  = (bf16_t)(o0[r] * inv);
    op[16] = (bf16_t)(o1[r] * inv);
    op[32] = (bf16_t)(o2[r] * inv);
    op[48] = (bf16_t)(o3[r] * inv);
  }
}

// ---------------------------------------------------------------------------
// Host-side launch
// ---------------------------------------------------------------------------
extern "C" void kernel_launch(void* const* d_in, const int* in_sizes, int n_in,
                              void* d_out, int out_size, void* d_ws, size_t ws_size,
                              hipStream_t stream) {
  const float* x     = (const float*)d_in[0];   // (B,S,DM)
  const float* w_qkv = (const float*)d_in[1];   // (DM,H,F)
  const float* b_qkv = (const float*)d_in[2];   // (H,F)
  const float* w_out = (const float*)d_in[3];   // (H,DH,DM)
  const float* b_out = (const float*)d_in[4];   // (DM,)
  (void)in_sizes; (void)n_in; (void)out_size; (void)ws_size;

  char* ws = (char*)d_ws;
  bf16_t* xb    = (bf16_t*)(ws + OFF_XB);
  bf16_t* wq_t  = (bf16_t*)(ws + OFF_WQT);
  bf16_t* wo_t  = (bf16_t*)(ws + OFF_WOT);
  bf16_t* qkvb  = (bf16_t*)(ws + OFF_QKV);
  bf16_t* qb    = (bf16_t*)(ws + OFF_Q);
  bf16_t* kb    = (bf16_t*)(ws + OFF_K);
  bf16_t* vtb   = (bf16_t*)(ws + OFF_VT);
  bf16_t* attnb = (bf16_t*)(ws + OFF_AT);

  // 1) pack operands to bf16 (weights pre-transposed for B-frag loads)
  {
    int n = M_ * DM_;
    cvt_f32_to_bf16<<<(n + 255) / 256, 256, 0, stream>>>(x, xb, n);
  }
  transpose_cvt_bf16<<<(DM_ * NQKV_ + 255) / 256, 256, 0, stream>>>(w_qkv, wq_t, DM_, NQKV_);
  transpose_cvt_bf16<<<(NAV_ * DM_ + 255) / 256, 256, 0, stream>>>(w_out, wo_t, NAV_, DM_);

  // 2) QKV projection: (4096 x 1024) x (1024 x 3072) + b_qkv -> bf16
  gemm_bf16_wmma_lds<<<dim3(NQKV_ / BN, M_ / BM), 512, 0, stream>>>(
      xb, wq_t, (void*)qkvb, b_qkv, M_, NQKV_, DM_, /*out_bf16=*/1);

  // 3) RoPE + split/scatter into attention-friendly layouts
  {
    int n = B_ * S_ * H_ * DH_;
    rope_scatter<<<(n + 255) / 256, 256, 0, stream>>>(qkvb, qb, kb, vtb);
  }

  // 4) causal flash attention with ALiBi (64 q rows / block, 4 waves)
  flash_attn_wmma<<<dim3(S_ / 64, H_, B_), 128, 0, stream>>>(qb, kb, vtb, attnb);

  // 5) output projection: (4096 x 1024) x (1024 x 1024) + b_out -> f32 out
  gemm_bf16_wmma_lds<<<dim3(DM_ / BN, M_ / BM), 512, 0, stream>>>(
      attnb, wo_t, d_out, b_out, M_, DM_, NAV_, /*out_bf16=*/0);
}
